// SetAbstraction_85993835200541
// MI455X (gfx1250) — matrix operand order repaired and checked by reading, not compiled, not run on hardware
//
#include <hip/hip_runtime.h>

#define BATCH   16
#define NPTS    4096
#define NPOINTS 1024
#define NSAMP   32
#define CFEAT   64
#define PROWS   (BATCH * NPOINTS * NSAMP)   // 524288 rows through the MLP
#define RT      (PROWS / 16)                // 32768 row tiles
#define BN_EPS  1e-5f

typedef __attribute__((ext_vector_type(16))) _Float16 v16h;
typedef __attribute__((ext_vector_type(8)))  _Float16 v8h;
typedef __attribute__((ext_vector_type(8)))  float    v8f;

// ---------------------------------------------------------------------------
// 1) Farthest point sampling: one block per batch, 1024 sequential steps.
//    Each thread owns 16 points (strided), running min-dist in registers,
//    block argmax via LDS tree (lower index wins ties, matching jnp.argmax).
// ---------------------------------------------------------------------------
__global__ void fps_kernel(const float* __restrict__ xyz, int* __restrict__ fpsIdx,
                           float* __restrict__ newXyzWs, float* __restrict__ outNewXyz) {
  const int b = blockIdx.x;
  const int t = threadIdx.x;
  const float* xb = xyz + (size_t)b * NPTS * 3;
  __shared__ float curX, curY, curZ;
  __shared__ float redD[256];
  __shared__ int   redI[256];
  float d[16];
#pragma unroll
  for (int j = 0; j < 16; ++j) d[j] = 1e10f;
  int far = 0;
  for (int i = 0; i < NPOINTS; ++i) {
    if (t == 0) {
      fpsIdx[b * NPOINTS + i] = far;
      float cx = xb[far * 3 + 0], cy = xb[far * 3 + 1], cz = xb[far * 3 + 2];
      curX = cx; curY = cy; curZ = cz;
      float* o = outNewXyz + ((size_t)b * NPOINTS + i) * 3;
      o[0] = cx; o[1] = cy; o[2] = cz;
      float* w = newXyzWs + ((size_t)b * NPOINTS + i) * 3;
      w[0] = cx; w[1] = cy; w[2] = cz;
    }
    __syncthreads();
    const float cx = curX, cy = curY, cz = curZ;
    float best = -1.0f; int bestI = 0;
#pragma unroll
    for (int j = 0; j < 16; ++j) {
      const int n = t + j * 256;
      float dx = xb[n * 3 + 0] - cx, dy = xb[n * 3 + 1] - cy, dz = xb[n * 3 + 2] - cz;
      float dd = dx * dx + dy * dy + dz * dz;
      d[j] = fminf(d[j], dd);
      if (d[j] > best) { best = d[j]; bestI = n; }
    }
    redD[t] = best; redI[t] = bestI;
    __syncthreads();
    for (int s = 128; s > 0; s >>= 1) {
      if (t < s) {
        float dv = redD[t + s]; int iv = redI[t + s];
        if (dv > redD[t] || (dv == redD[t] && iv < redI[t])) { redD[t] = dv; redI[t] = iv; }
      }
      __syncthreads();
    }
    far = redI[0];
    __syncthreads();
  }
}

// ---------------------------------------------------------------------------
// 2) Transpose features (B,C,N) -> (B,N,C) in f16 so per-point gathers are
//    two contiguous 64B reads.
// ---------------------------------------------------------------------------
__global__ void feat_transpose_kernel(const float* __restrict__ feat,
                                      _Float16* __restrict__ featT) {
  size_t i = (size_t)blockIdx.x * blockDim.x + threadIdx.x;
  const size_t total = (size_t)BATCH * NPTS * CFEAT;
  if (i >= total) return;
  int c = (int)(i % CFEAT);
  int n = (int)((i / CFEAT) % NPTS);
  int b = (int)(i / ((size_t)CFEAT * NPTS));
  featT[i] = (_Float16)feat[((size_t)b * CFEAT + c) * NPTS + n];
}

// ---------------------------------------------------------------------------
// 3) kNN: one wave per centroid (4 waves/block, 64KB LDS of dists).
//    Each lane privately owns slots n == lane (mod 32) -> no barriers needed.
//    32 rounds of wave argmin + winner-lane rescan.
// ---------------------------------------------------------------------------
__global__ void knn_kernel(const float* __restrict__ xyz, const float* __restrict__ newXyz,
                           int* __restrict__ knnIdx) {
  __shared__ float distS[4 * NPTS];
  const int lane = threadIdx.x & 31;
  const int w    = threadIdx.x >> 5;
  const int gid  = blockIdx.x * 4 + w;     // centroid id = b*1024 + m
  const int b    = gid >> 10;
  const float* xb = xyz + (size_t)b * NPTS * 3;
  const float cx = newXyz[gid * 3 + 0], cy = newXyz[gid * 3 + 1], cz = newXyz[gid * 3 + 2];
  float* ds = distS + w * NPTS;
  float lmin = 3.0e38f; int lidx = 0;
  for (int j = 0; j < NPTS / 32; ++j) {
    const int n = lane + j * 32;
    float dx = xb[n * 3 + 0] - cx, dy = xb[n * 3 + 1] - cy, dz = xb[n * 3 + 2] - cz;
    float dd = dx * dx + dy * dy + dz * dz;
    ds[n] = dd;
    if (dd < lmin) { lmin = dd; lidx = n; }
  }
  for (int r = 0; r < NSAMP; ++r) {
    float dv = lmin; int iv = lidx;
    for (int off = 16; off > 0; off >>= 1) {
      float od = __shfl_xor(dv, off);
      int   oi = __shfl_xor(iv, off);
      if (od < dv || (od == dv && oi < iv)) { dv = od; iv = oi; }
    }
    if (lane == 0) knnIdx[gid * NSAMP + r] = iv;
    if ((iv & 31) == lane) {
      ds[iv] = 3.0e38f;
      lmin = 3.0e38f; lidx = 0;
      for (int j = 0; j < NPTS / 32; ++j) {
        const int n = lane + j * 32;
        float dd = ds[n];
        if (dd < lmin) { lmin = dd; lidx = n; }
      }
    }
  }
}

// ---------------------------------------------------------------------------
// 4) Pack weights (cin x cout, f32) into per-lane WMMA B fragments (f16),
//    K padded to multiple of 32. Fragment layout (ISA 7.12.2):
//    lane: N = ct*16 + (lane&15), K = kc*32 + (h<8 ? (lane>>4)*8 + h : 16 + (lane>>4)*8 + h-8), h=0..15.
// ---------------------------------------------------------------------------
__global__ void pack_w_kernel(const float* __restrict__ W, _Float16* __restrict__ Bp,
                              int cin, int cout) {
  int i = blockIdx.x * blockDim.x + threadIdx.x;
  const int KC = (cin + 31) >> 5;
  const int CT = cout >> 4;
  const int total = KC * CT * 32 * 16;
  if (i >= total) return;
  int h    = i & 15;
  int lane = (i >> 4) & 31;
  int ct   = (i >> 9) % CT;
  int kc   = i / (CT * 512);
  int K = kc * 32 + ((h >> 3) * 16) + ((lane >> 4) * 8) + (h & 7);
  int N = ct * 16 + (lane & 15);
  float v = (K < cin) ? W[K * cout + N] : 0.0f;
  Bp[i] = (_Float16)v;
}

// ---------------------------------------------------------------------------
// 5) Layer-1 GEMM with fused grouping: 4 row tiles per block, ping-pong LDS
//    A-stage (16 x 96 halves, 67 real + pad), B fragments held in registers,
//    4 waves = 4 col tiles of W0. All A fragments loaded before the WMMA
//    chain so one wait covers all ds_loads. Stats accumulate across tiles,
//    one atomic pair per lane at the end.
// ---------------------------------------------------------------------------
#define TPB1 4
__global__ void gemm1_kernel(const float* __restrict__ xyz, const _Float16* __restrict__ featT,
                             const int* __restrict__ knnIdx, const float* __restrict__ newXyz,
                             const _Float16* __restrict__ Bp, const float* __restrict__ bias,
                             _Float16* __restrict__ out, float* __restrict__ stats) {
  __shared__ alignas(32) _Float16 sA[2][16 * 96];
  const int t    = threadIdx.x;
  const int lane = t & 31;
  const int ct   = t >> 5;       // wave id == output col tile (0..3)
  const int r    = lane & 15;
  const int hi   = lane >> 4;
  // hoist the 3 B fragments for this wave's col tile
  v16h bf[3];
#pragma unroll
  for (int kc = 0; kc < 3; ++kc)
    bf[kc] = *(const v16h*)(Bp + ((size_t)(kc * 4 + ct) * 32 + lane) * 16);
  const int   col = ct * 16 + r;
  const float bc  = bias[col];
  float s1 = 0.0f, s2 = 0.0f;

  const int row = t >> 3;        // 0..15 (gather role)
  const int seg = t & 7;         // 0..7, 12 halves each
  auto stage = [&](int it, int buf) {
    const int grow = (blockIdx.x * TPB1 + it) * 16 + row;  // ((b*M+m)*K + k)
    const int k = grow & 31;
    const int g = grow >> 5;                                // b*1024+m
    const int b = grow >> 15;
    const int pidx = knnIdx[g * NSAMP + k];
    const float cx = newXyz[g * 3 + 0], cy = newXyz[g * 3 + 1], cz = newXyz[g * 3 + 2];
    const float* p = xyz + ((size_t)b * NPTS + pidx) * 3;
    const _Float16* f = featT + ((size_t)b * NPTS + pidx) * CFEAT;
#pragma unroll
    for (int q = 0; q < 12; ++q) {
      const int c = seg * 12 + q;
      float v;
      if (c < 3)       v = p[c] - (c == 0 ? cx : (c == 1 ? cy : cz));
      else if (c < 67) v = (float)f[c - 3];
      else             v = 0.0f;
      sA[buf][row * 96 + c] = (_Float16)v;
    }
  };

  stage(0, 0);
  __syncthreads();
  for (int it = 0; it < TPB1; ++it) {
    if (it + 1 < TPB1) stage(it + 1, (it + 1) & 1);
    const _Float16* sa = sA[it & 1];
    // load ALL A fragments first so the ds_loads pipeline under one wait
    v16h a[3];
#pragma unroll
    for (int kc = 0; kc < 3; ++kc) {
      v8h lo = *(const v8h*)(sa + r * 96 + kc * 32 + hi * 8);
      v8h hh = *(const v8h*)(sa + r * 96 + kc * 32 + 16 + hi * 8);
#pragma unroll
      for (int q = 0; q < 8; ++q) { a[kc][q] = lo[q]; a[kc][q + 8] = hh[q]; }
    }
    v8f acc = {};
#pragma unroll
    for (int kc = 0; kc < 3; ++kc)
      acc = __builtin_amdgcn_wmma_f32_16x16x32_f16(false, a[kc], false, bf[kc],
                                                   (short)0, acc, false, false);
    const size_t rbase = ((size_t)blockIdx.x * TPB1 + it) * 16;
#pragma unroll
    for (int j = 0; j < 8; ++j) {
      float v = acc[j] + bc;
      out[(rbase + (hi << 3) + j) * 64 + col] = (_Float16)v;
      s1 += v; s2 += v * v;
    }
    __syncthreads();
  }
  s1 += __shfl_xor(s1, 16);
  s2 += __shfl_xor(s2, 16);
  if (hi == 0) {
    atomicAdd(&stats[col], s1);
    atomicAdd(&stats[64 + col], s2);
  }
}

// ---------------------------------------------------------------------------
// 6) Layers 2/3 GEMM: 8 row tiles per block, ping-pong LDS A-stage
//    (16 x 64 halves, b128 coalesced), CT waves cover CT col tiles,
//    B fragments in registers, A fragments loaded ahead of the WMMA chain,
//    stats accumulated across tiles.
// ---------------------------------------------------------------------------
#define TPB23 8
template <int CT>
__global__ void gemm23_kernel(const _Float16* __restrict__ A, const _Float16* __restrict__ Bp,
                              const float* __restrict__ bias, _Float16* __restrict__ out,
                              float* __restrict__ stats) {
  constexpr int COUT = CT * 16;
  __shared__ alignas(32) _Float16 sA[2][16 * 64];
  const int t    = threadIdx.x;
  const int lane = t & 31;
  const int ct   = t >> 5;
  const int r    = lane & 15;
  const int hi   = lane >> 4;
  v16h bf[2];
#pragma unroll
  for (int kc = 0; kc < 2; ++kc)
    bf[kc] = *(const v16h*)(Bp + ((size_t)(kc * CT + ct) * 32 + lane) * 16);
  const int   col = ct * 16 + r;
  const float bc  = bias[col];
  float s1 = 0.0f, s2 = 0.0f;

  auto stage = [&](int it, int buf) {
    if (t < 128) {
      const int row = t >> 3, seg = t & 7;
      *(v8h*)(sA[buf] + row * 64 + seg * 8) =
          *(const v8h*)(A + (((size_t)blockIdx.x * TPB23 + it) * 16 + row) * 64 + seg * 8);
    }
  };

  stage(0, 0);
  __syncthreads();
  for (int it = 0; it < TPB23; ++it) {
    if (it + 1 < TPB23) stage(it + 1, (it + 1) & 1);
    const _Float16* sa = sA[it & 1];
    v16h a[2];
#pragma unroll
    for (int kc = 0; kc < 2; ++kc) {
      v8h lo = *(const v8h*)(sa + r * 64 + kc * 32 + hi * 8);
      v8h hh = *(const v8h*)(sa + r * 64 + kc * 32 + 16 + hi * 8);
#pragma unroll
      for (int q = 0; q < 8; ++q) { a[kc][q] = lo[q]; a[kc][q + 8] = hh[q]; }
    }
    v8f acc = {};
#pragma unroll
    for (int kc = 0; kc < 2; ++kc)
      acc = __builtin_amdgcn_wmma_f32_16x16x32_f16(false, a[kc], false, bf[kc],
                                                   (short)0, acc, false, false);
    const size_t rbase = ((size_t)blockIdx.x * TPB23 + it) * 16;
#pragma unroll
    for (int j = 0; j < 8; ++j) {
      float v = acc[j] + bc;
      out[(rbase + (hi << 3) + j) * COUT + col] = (_Float16)v;
      s1 += v; s2 += v * v;
    }
    __syncthreads();
  }
  s1 += __shfl_xor(s1, 16);
  s2 += __shfl_xor(s2, 16);
  if (hi == 0) {
    atomicAdd(&stats[col], s1);
    atomicAdd(&stats[COUT + col], s2);
  }
}

// ---------------------------------------------------------------------------
// 7) In-place training-mode BN + ReLU. Per-channel scale/shift precomputed
//    into LDS once per block; each thread processes one contiguous v8h
//    (b128 load/store). stats = [sum[C], sumsq[C]].
// ---------------------------------------------------------------------------
__global__ void bnrelu_kernel(_Float16* __restrict__ act, const float* __restrict__ stats,
                              const float* __restrict__ g, const float* __restrict__ beta,
                              int C) {
  __shared__ float sSc[128], sSh[128];
  const int t = threadIdx.x;
  if (t < C) {
    const float invP = 1.0f / (float)PROWS;
    const float mu  = stats[t] * invP;
    const float var = stats[C + t] * invP - mu * mu;
    const float sc  = g[t] * rsqrtf(var + BN_EPS);
    sSc[t] = sc;
    sSh[t] = beta[t] - mu * sc;
  }
  __syncthreads();
  const size_t base = ((size_t)blockIdx.x * 256 + t) * 8;   // grid sized exactly
  const int c0 = (int)(base % (size_t)C);                   // multiple of 8, no wrap
  v8h x = *(const v8h*)(act + base);
  v8h y;
#pragma unroll
  for (int q = 0; q < 8; ++q) {
    float v = (float)x[q] * sSc[c0 + q] + sSh[c0 + q];
    y[q] = (_Float16)fmaxf(v, 0.0f);
  }
  *(v8h*)(act + base) = y;
}

// ---------------------------------------------------------------------------
// 8) Layer-3 BN + ReLU fused with max-pool over K=32, write (B,128,M) fp32.
// ---------------------------------------------------------------------------
__global__ void bn_maxpool_kernel(const _Float16* __restrict__ act3,
                                  const float* __restrict__ stats,
                                  const float* __restrict__ g, const float* __restrict__ beta,
                                  float* __restrict__ outF) {
  const int gId = blockIdx.x;         // b*1024 + m
  const int c   = threadIdx.x;        // 0..127
  const float invP = 1.0f / (float)PROWS;
  const float mu  = stats[c] * invP;
  const float var = stats[128 + c] * invP - mu * mu;
  const float sc  = g[c] * rsqrtf(var + BN_EPS);
  const float sh  = beta[c] - mu * sc;
  float m = 0.0f;                      // ReLU outputs are >= 0
  const _Float16* p = act3 + (size_t)gId * NSAMP * 128 + c;
#pragma unroll
  for (int k = 0; k < NSAMP; ++k) {
    float v = (float)p[k * 128] * sc + sh;
    m = fmaxf(m, fmaxf(v, 0.0f));
  }
  const int b = gId >> 10, mm = gId & 1023;
  outF[((size_t)b * 128 + c) * NPOINTS + mm] = m;
}

// ---------------------------------------------------------------------------
extern "C" void kernel_launch(void* const* d_in, const int* in_sizes, int n_in,
                              void* d_out, int out_size, void* d_ws, size_t ws_size,
                              hipStream_t stream) {
  const float* xyz      = (const float*)d_in[0];
  const float* features = (const float*)d_in[1];
  const float* W0 = (const float*)d_in[2];  const float* b0 = (const float*)d_in[3];
  const float* g0 = (const float*)d_in[4];  const float* e0 = (const float*)d_in[5];
  const float* W1 = (const float*)d_in[6];  const float* b1 = (const float*)d_in[7];
  const float* g1 = (const float*)d_in[8];  const float* e1 = (const float*)d_in[9];
  const float* W2 = (const float*)d_in[10]; const float* b2 = (const float*)d_in[11];
  const float* g2 = (const float*)d_in[12]; const float* e2 = (const float*)d_in[13];

  float* outNewXyz = (float*)d_out;
  float* outFeat   = (float*)d_out + (size_t)BATCH * NPOINTS * 3;

  // workspace carve-out (256B aligned)
  char* ws = (char*)d_ws;
  size_t off = 0;
  auto take = [&](size_t bytes) -> void* {
    void* p = ws + off;
    off = (off + bytes + 255) & ~(size_t)255;
    return p;
  };
  int*      fpsIdx   = (int*)take((size_t)BATCH * NPOINTS * 4);
  int*      knnIdx   = (int*)take((size_t)BATCH * NPOINTS * NSAMP * 4);
  float*    newXyzWs = (float*)take((size_t)BATCH * NPOINTS * 3 * 4);
  float*    stats    = (float*)take(2048);          // [0:128)=L1, [128:256)=L2, [256:512)=L3
  _Float16* Bp0      = (_Float16*)take(3 * 4 * 32 * 16 * 2);
  _Float16* Bp1      = (_Float16*)take(2 * 4 * 32 * 16 * 2);
  _Float16* Bp2      = (_Float16*)take(2 * 8 * 32 * 16 * 2);
  _Float16* featT    = (_Float16*)take((size_t)BATCH * NPTS * CFEAT * 2);
  _Float16* act1     = (_Float16*)take((size_t)PROWS * 64 * 2);
  _Float16* act2     = (_Float16*)take((size_t)PROWS * 64 * 2);
  _Float16* act3     = (_Float16*)take((size_t)PROWS * 128 * 2);
  float* stats1 = stats;
  float* stats2 = stats + 128;
  float* stats3 = stats + 256;

  hipMemsetAsync(stats, 0, 2048, stream);

  fps_kernel<<<BATCH, 256, 0, stream>>>(xyz, fpsIdx, newXyzWs, outNewXyz);
  feat_transpose_kernel<<<(BATCH * NPTS * CFEAT) / 256, 256, 0, stream>>>(features, featT);
  knn_kernel<<<(BATCH * NPOINTS) / 4, 128, 0, stream>>>(xyz, newXyzWs, knnIdx);
  pack_w_kernel<<<(3 * 4 * 32 * 16 + 255) / 256, 256, 0, stream>>>(W0, Bp0, 67, 64);
  pack_w_kernel<<<(2 * 4 * 32 * 16 + 255) / 256, 256, 0, stream>>>(W1, Bp1, 64, 64);
  pack_w_kernel<<<(2 * 8 * 32 * 16 + 255) / 256, 256, 0, stream>>>(W2, Bp2, 64, 128);

  gemm1_kernel<<<RT / TPB1, 128, 0, stream>>>(xyz, featT, knnIdx, newXyzWs, Bp0, b0, act1, stats1);
  bnrelu_kernel<<<(PROWS * 64) / (256 * 8), 256, 0, stream>>>(act1, stats1, g0, e0, 64);
  gemm23_kernel<4><<<RT / TPB23, 128, 0, stream>>>(act1, Bp1, b1, act2, stats2);
  bnrelu_kernel<<<(PROWS * 64) / (256 * 8), 256, 0, stream>>>(act2, stats2, g1, e1, 64);
  gemm23_kernel<8><<<RT / TPB23, 256, 0, stream>>>(act2, Bp2, b2, act3, stats3);
  bn_maxpool_kernel<<<BATCH * NPOINTS, 128, 0, stream>>>(act3, stats3, g2, e2, outFeat);
}
